// EmergentGatedFFN_20547123544590
// MI455X (gfx1250) — matrix operand order repaired and checked by reading, not compiled, not run on hardware
//
#include <hip/hip_runtime.h>

typedef __attribute__((ext_vector_type(16))) __bf16 v16bf;
typedef __attribute__((ext_vector_type(8)))  __bf16 v8bf;
typedef __attribute__((ext_vector_type(8)))  float  v8f;
typedef __attribute__((ext_vector_type(4)))  int    v4i;

#define D_MODEL 1024
#define D_FF    4096
#define NTILES  4
#define TILE_UP 1024
#define TILE_DN 256

#define BM 128          // tokens per block (8 M-tiles, one per wave)
#define FK 32           // f-chunk width (one bf16 WMMA K-step for GEMM2)
#define KP 128          // K staging piece for w_up

// LDS row strides (bf16 elements), padded for 16B alignment + bank spread
#define XS  1032        // x tile rows [BM][XS]
#define BS1 136         // w_up piece  [FK][BS1]
#define BS2 40          // w_down chunk[256][BS2]
#define HS  40          // hidden chunk[BM][HS]

#define X_BYTES   (BM * XS * 2)       // 264192
#define B1_BYTES  (FK * BS1 * 2)      // 8704 (x2 buffers)
#define B2_BYTES  (256 * BS2 * 2)     // 20480
#define H_BYTES   (BM * HS * 2)       // 10240
#define BUP_BYTES (FK * 4)            // 128
#define BDN_BYTES (256 * 4)           // 1024
#define TOK_BYTES (BM * 4)            // 512
#define SMEM_BYTES (X_BYTES + 2*B1_BYTES + B2_BYTES + H_BYTES + BUP_BYTES + BDN_BYTES + TOK_BYTES) // 313984

// ------------------------- async global->LDS helpers ------------------------
#if defined(__gfx1250__) && __has_builtin(__builtin_amdgcn_global_load_async_to_lds_b128)
#define HAVE_ASYNC 1
#else
#define HAVE_ASYNC 0
#endif

#if HAVE_ASYNC
typedef __attribute__((address_space(1))) v4i v4i_glob;
typedef __attribute__((address_space(3))) v4i v4i_lds;
#if __has_builtin(__builtin_amdgcn_s_wait_asynccnt)
#define WAIT_ASYNC(n) __builtin_amdgcn_s_wait_asynccnt(n)
#else
#define WAIT_ASYNC(n) asm volatile("s_wait_asynccnt " #n ::: "memory")
#endif
__device__ __forceinline__ void async_cp16(const __bf16* g, __bf16* l) {
  __builtin_amdgcn_global_load_async_to_lds_b128((v4i_glob*)g, (v4i_lds*)l, 0, 0);
}
#else
#define WAIT_ASYNC(n) do {} while (0)
__device__ __forceinline__ void async_cp16(const __bf16* g, __bf16* l) {
  *(uint4*)l = *(const uint4*)g;   // sync fallback; barrier provides visibility
}
#endif

// ---------------------------------------------------------------------------
// 1) signatures[t][c] = sign(sum_r w_up[t*1024+r][c])
__global__ __launch_bounds__(256) void sig_kernel(const float* __restrict__ wup,
                                                  float* __restrict__ sig) {
  int idx = blockIdx.x * 256 + threadIdx.x;        // 0..4095
  int t = idx >> 10, c = idx & 1023;
  const float* p = wup + (long)t * TILE_UP * D_MODEL + c;
  float s = 0.f;
  for (int r = 0; r < TILE_UP; ++r) s += p[(long)r * D_MODEL];
  sig[idx] = (s > 0.f) ? 1.f : (s < 0.f ? -1.f : 0.f);
}

// 2a) w_up f32 -> bf16 (full matrix, row-major [4096][1024])
__global__ __launch_bounds__(256) void cvt_wup(const float* __restrict__ w,
                                               __bf16* __restrict__ dst) {
  long i = (long)blockIdx.x * 1024 + (long)threadIdx.x * 4;
  float4 f = *(const float4*)(w + i);
  __bf16* d = dst + i;
  d[0] = (__bf16)f.x; d[1] = (__bf16)f.y; d[2] = (__bf16)f.z; d[3] = (__bf16)f.w;
}

// 2b) live w_down blocks -> bf16 compact [e][256][1024]
__global__ __launch_bounds__(256) void cvt_wdown(const float* __restrict__ w,
                                                 __bf16* __restrict__ dst) {
  long i = (long)blockIdx.x * 1024 + (long)threadIdx.x * 4;  // over 4*256*1024
  int e = (int)(i >> 18);
  int r = (int)((i >> 10) & 255);
  int c = (int)(i & 1023);
  const float* src = w + ((long)(e * TILE_DN + r)) * D_FF + e * TILE_UP + c;
  float4 f = *(const float4*)src;
  __bf16* d = dst + i;
  d[0] = (__bf16)f.x; d[1] = (__bf16)f.y; d[2] = (__bf16)f.z; d[3] = (__bf16)f.w;
}

// 3) routing + x bf16 conversion: one wave32 per token
__global__ __launch_bounds__(256) void route_cvt_kernel(const float* __restrict__ x,
                                                        const float* __restrict__ sig,
                                                        __bf16* __restrict__ xb,
                                                        float* __restrict__ gate_out,
                                                        int* __restrict__ lists,
                                                        int* __restrict__ counts, int N) {
  __shared__ float sl[NTILES * D_MODEL];
  for (int i = threadIdx.x; i < NTILES * D_MODEL; i += 256) sl[i] = sig[i];
  __syncthreads();

  int token = blockIdx.x * 8 + (threadIdx.x >> 5);
  int lane = threadIdx.x & 31;
  if (token >= N) return;
  const float* xr = x + (long)token * D_MODEL;
  __bf16* xw = xb + (long)token * D_MODEL;
  float s0 = 0.f, s1 = 0.f, s2 = 0.f, s3 = 0.f;
  for (int i = lane; i < D_MODEL; i += 32) {
    float xv = xr[i];
    xw[i] = (__bf16)xv;
    s0 += xv * sl[i];
    s1 += xv * sl[1024 + i];
    s2 += xv * sl[2048 + i];
    s3 += xv * sl[3072 + i];
  }
  for (int off = 16; off; off >>= 1) {
    s0 += __shfl_xor(s0, off, 32);
    s1 += __shfl_xor(s1, off, 32);
    s2 += __shfl_xor(s2, off, 32);
    s3 += __shfl_xor(s3, off, 32);
  }
  if (lane == 0) {
    float sc[4] = {s0, s1, s2, s3};
    int w = 0; float best = sc[0];
    for (int t = 1; t < 4; ++t) if (sc[t] > best) { best = sc[t]; w = t; }
    for (int t = 0; t < 4; ++t) gate_out[(long)token * 4 + t] = (t == w) ? 1.f : 0.f;
    int slot = atomicAdd(&counts[w], 1);
    lists[(long)w * N + slot] = token;
  }
}

// --------------------- WMMA fragment loaders (from LDS) ---------------------
// A-fragment (16-bit, 16x32): lanes 0-15 K=0..15, lanes 16-31 K=16..31
__device__ __forceinline__ v16bf load_fragA(const __bf16* base, int row, int stride, int klo) {
  union { v16bf v; v8bf h[2]; } u;
  const __bf16* p = base + row * stride + klo;
  u.h[0] = *(const v8bf*)p;
  u.h[1] = *(const v8bf*)(p + 8);
  return u.v;
}
// B-fragment (32x16): lane n<16 holds K=0..15 of column n; lanes 16-31 hold K=16..31
__device__ __forceinline__ v16bf load_fragB(const __bf16* base, int row, int stride, int k) {
  union { v16bf v; v8bf h[2]; } u;
  const __bf16* p = base + row * stride + k;
  u.h[0] = *(const v8bf*)p;
  u.h[1] = *(const v8bf*)(p + 8);
  return u.v;
}

// 4) grouped-expert fused FFN: GEMM1 (up, ReLU) streamed into GEMM2 (down)
__global__ __launch_bounds__(256) void expert_gemm(const __bf16* __restrict__ xb,   // [N][1024] bf16
                                                   const __bf16* __restrict__ wup,  // [4096][1024] bf16
                                                   const __bf16* __restrict__ wdc,  // [4][256][1024] bf16
                                                   const float* __restrict__ b_up,
                                                   const float* __restrict__ b_down,
                                                   const int* __restrict__ lists,
                                                   const int* __restrict__ counts,
                                                   float* __restrict__ out, int N) {
  extern __shared__ char smem[];
  __bf16* xs   = (__bf16*)smem;                                       // [BM][XS]
  __bf16* b1a  = (__bf16*)(smem + X_BYTES);                           // [FK][BS1]
  __bf16* b1b  = (__bf16*)(smem + X_BYTES + B1_BYTES);                // [FK][BS1]
  __bf16* b2   = (__bf16*)(smem + X_BYTES + 2*B1_BYTES);              // [256][BS2]
  __bf16* hs   = (__bf16*)(smem + X_BYTES + 2*B1_BYTES + B2_BYTES);   // [BM][HS]
  float* bup_s = (float*)(smem + X_BYTES + 2*B1_BYTES + B2_BYTES + H_BYTES);
  float* bdn_s = (float*)((char*)bup_s + BUP_BYTES);
  int*   tok_s = (int*)((char*)bdn_s + BDN_BYTES);

  const int e = blockIdx.y;
  const int mb = blockIdx.x;
  const int cnt = counts[e];
  if (mb * BM >= cnt) return;

  const int tid = threadIdx.x;
  const int wave = tid >> 5;
  const int lane = tid & 31;
  const int hi = lane >> 4;        // which 16-lane half
  const int l16 = lane & 15;
  const int mrow0 = wave * 16;     // this wave's token-row base in the block

  // stage token ids / down bias
  if (tid < BM) {
    int slot = mb * BM + tid;
    tok_s[tid] = (slot < cnt) ? lists[(long)e * N + slot] : -1;
  }
  bdn_s[tid] = b_down[e * TILE_DN + tid];
  __syncthreads();

  // stage x rows (gathered, bf16) via pipelined async copies: 64 b128 ops/thread
  for (int j = 0; j < 64; ++j) {
    int o = tid + 256 * j;                 // 0..16383
    int r = o >> 7;                        // token row in block
    int cb = (o & 127) * 8;                // bf16 column (16B granules)
    int t = tok_s[r];
    long tt = (t < 0) ? 0 : t;
    async_cp16(xb + tt * D_MODEL + cb, xs + r * XS + cb);
  }
  WAIT_ASYNC(0);
  __syncthreads();

  v8f acc2[16];
  #pragma unroll
  for (int i = 0; i < 16; ++i) acc2[i] = v8f{};

  const int wr = tid >> 3;                 // w_up staging: row within FK
  const int wc = (tid & 7) * 16;           // w_up staging: col within KP

  for (int fc = 0; fc < TILE_UP; fc += FK) {
    // stage per-chunk up-bias
    if (tid < FK) bup_s[tid] = b_up[e * TILE_UP + fc + tid];
    // issue async w_down chunk (completes, in order, before piece0 wait)
    {
      const __bf16* src = wdc + ((long)e * TILE_DN + tid) * TILE_UP + fc; // 64B contiguous
      __bf16* dst = b2 + tid * BS2;
      async_cp16(src, dst);
      async_cp16(src + 8, dst + 8);
      async_cp16(src + 16, dst + 16);
      async_cp16(src + 24, dst + 24);
    }
    // issue w_up piece 0
    {
      const __bf16* src = wup + ((long)(e * TILE_UP + fc + wr)) * D_MODEL + wc;
      async_cp16(src, b1a + wr * BS1 + wc);
      async_cp16(src + 8, b1a + wr * BS1 + wc + 8);
    }

    // ---------------- GEMM1: h[BM][FK] = x @ w_up_e[fc:fc+FK]^T ----------------
    v8f acc1[2];
    acc1[0] = v8f{}; acc1[1] = v8f{};

    for (int kp = 0; kp < 8; ++kp) {               // K pieces of KP=128
      __bf16* cur = (kp & 1) ? b1b : b1a;
      if (kp + 1 < 8) {                            // prefetch next piece
        __bf16* nxt = (kp & 1) ? b1a : b1b;
        const __bf16* src = wup + ((long)(e * TILE_UP + fc + wr)) * D_MODEL
                                + (kp + 1) * KP + wc;
        async_cp16(src, nxt + wr * BS1 + wc);
        async_cp16(src + 8, nxt + wr * BS1 + wc + 8);
        WAIT_ASYNC(2);                             // current piece (and b2) resident
      } else {
        WAIT_ASYNC(0);
      }
      __syncthreads();                             // piece visible to all waves
      #pragma unroll
      for (int ks = 0; ks < KP; ks += 32) {
        v16bf A  = load_fragA(xs, mrow0 + l16, XS, kp * KP + ks + hi * 16);
        v16bf B0 = load_fragB(cur, l16,      BS1, ks + hi * 16);
        v16bf Bb = load_fragB(cur, 16 + l16, BS1, ks + hi * 16);
        acc1[0] = __builtin_amdgcn_wmma_f32_16x16x32_bf16(false, A, false, B0,
                                                          (short)0, acc1[0], false, false);
        acc1[1] = __builtin_amdgcn_wmma_f32_16x16x32_bf16(false, A, false, Bb,
                                                          (short)0, acc1[1], false, false);
      }
      __syncthreads();                             // compute done before buffer reuse
    }

    // bias + ReLU, write bf16 hidden chunk (own wave's rows only; DS in-order per wave)
    #pragma unroll
    for (int nt = 0; nt < 2; ++nt) {
      int n = nt * 16 + l16;
      float bb = bup_s[n];
      #pragma unroll
      for (int v = 0; v < 8; ++v) {
        float hv = acc1[nt][v] + bb;
        hv = hv > 0.f ? hv : 0.f;
        hs[(mrow0 + hi * 8 + v) * HS + n] = (__bf16)hv;
      }
    }

    // ---------------- GEMM2: out[BM][256] += h_chunk @ w_down_chunk^T ----------------
    v16bf A2 = load_fragA(hs, mrow0 + l16, HS, hi * 16);
    #pragma unroll
    for (int nt = 0; nt < 16; ++nt) {
      v16bf Bf = load_fragB(b2, nt * 16 + l16, BS2, hi * 16);
      acc2[nt] = __builtin_amdgcn_wmma_f32_16x16x32_bf16(false, A2, false, Bf,
                                                         (short)0, acc2[nt], false, false);
    }
    __syncthreads();                               // b2/bup reads done before restage
  }

  // epilogue: winner slice (+bias), zero the gated-out 768 columns (bias gated too)
  #pragma unroll
  for (int nt = 0; nt < 16; ++nt) {
    int d = nt * 16 + l16;
    float bb = bdn_s[d];
    #pragma unroll
    for (int v = 0; v < 8; ++v) {
      int m = hi * 8 + v;
      int t = tok_s[mrow0 + m];
      if (t >= 0) out[(long)t * D_MODEL + e * TILE_DN + d] = acc2[nt][v] + bb;
    }
  }
  const int e64 = e * 64;  // float4 index where the winner slice starts
  for (int idx = tid; idx < BM * 192; idx += 256) {
    int r = idx / 192, c4 = idx % 192;
    int t = tok_s[r];
    if (t >= 0) {
      int col4 = (c4 < e64) ? c4 : c4 + 64;
      ((float4*)(out + (long)t * D_MODEL))[col4] = make_float4(0.f, 0.f, 0.f, 0.f);
    }
  }
}

// ---------------------------------------------------------------------------
extern "C" void kernel_launch(void* const* d_in, const int* in_sizes, int n_in,
                              void* d_out, int out_size, void* d_ws, size_t ws_size,
                              hipStream_t stream) {
  const float* x      = (const float*)d_in[0];
  const float* w_up   = (const float*)d_in[1];
  const float* b_up   = (const float*)d_in[2];
  const float* w_down = (const float*)d_in[3];
  const float* b_down = (const float*)d_in[4];

  const int N = in_sizes[0] / D_MODEL;   // 16384 tokens
  float* out      = (float*)d_out;                     // [N][1024]
  float* gate_out = out + (size_t)N * D_MODEL;         // [N][4]

  char* ws = (char*)d_ws;
  size_t off = 0;
  float* sig    = (float*)(ws + off); off += (size_t)NTILES * D_MODEL * 4;   // 16KB
  int*   counts = (int*)(ws + off);   off += 256;
  int*   lists  = (int*)(ws + off);   off += (size_t)NTILES * N * 4;
  off = (off + 511) & ~(size_t)511;
  __bf16* wupb = (__bf16*)(ws + off); off += (size_t)D_FF * D_MODEL * 2;     // 8MB
  __bf16* wdcb = (__bf16*)(ws + off); off += (size_t)NTILES * TILE_DN * TILE_UP * 2; // 2MB
  __bf16* xb   = (__bf16*)(ws + off); off += (size_t)N * D_MODEL * 2;        // 32MB

  (void)hipMemsetAsync(counts, 0, NTILES * sizeof(int), stream);

  sig_kernel<<<(NTILES * D_MODEL) / 256, 256, 0, stream>>>(w_up, sig);
  cvt_wup<<<(D_FF * D_MODEL) / 1024, 256, 0, stream>>>(w_up, wupb);
  cvt_wdown<<<(NTILES * TILE_DN * TILE_UP) / 1024, 256, 0, stream>>>(w_down, wdcb);
  route_cvt_kernel<<<(N + 7) / 8, 256, 0, stream>>>(x, sig, xb, gate_out, lists, counts, N);

  dim3 grid((N + BM - 1) / BM, NTILES);
  expert_gemm<<<grid, 256, SMEM_BYTES, stream>>>(xb, wupb, wdcb, b_up, b_down,
                                                 lists, counts, out, N);
}
